// RWKV_TimeMix_24764781429091
// MI455X (gfx1250) — matrix operand order repaired and checked by reading, not compiled, not run on hardware
//
#include <hip/hip_runtime.h>

// Problem dims (fixed by the reference)
#define Bn 16384
#define Hn 1024
#define BHn ((size_t)Bn * (size_t)Hn)

typedef __bf16 bf16x8  __attribute__((ext_vector_type(8)));
typedef __bf16 bf16x16 __attribute__((ext_vector_type(16)));
typedef float  f32x8   __attribute__((ext_vector_type(8)));

#if defined(__AMDGCN__) && __has_builtin(__builtin_amdgcn_global_load_async_to_lds_b128) && \
    __has_builtin(__builtin_amdgcn_s_wait_asynccnt)
#define USE_ASYNC_LDS 1
#else
#define USE_ASYNC_LDS 0
#endif

// The async builtin takes pointers to 16B int vectors: AS1 source, AS3 dest.
typedef int i32x4v __attribute__((vector_size(16)));
typedef __attribute__((address_space(1))) i32x4v* gbl_v4p;
typedef __attribute__((address_space(3))) i32x4v* lds_v4p;

// 16-byte global -> LDS copy. Async (ASYNCcnt-tracked, no VGPR round trip)
// when the gfx1250 builtin is available; synchronous copy otherwise.
static __device__ __forceinline__ void copy16_g2l(const unsigned short* g,
                                                  unsigned short* l) {
#if USE_ASYNC_LDS
  // flat->global addrspacecast is value-preserving; flat->LDS is a 32-bit trunc.
  gbl_v4p gp = (gbl_v4p)(unsigned long long)(const void*)g;
  lds_v4p lp = (lds_v4p)(unsigned int)(unsigned long long)(void*)l;
  __builtin_amdgcn_global_load_async_to_lds_b128(gp, lp, 0, 0);
#else
  *reinterpret_cast<uint4*>(l) = *reinterpret_cast<const uint4*>(g);
#endif
}

template <int N>
static __device__ __forceinline__ void async_wait() {
#if USE_ASYNC_LDS
  __builtin_amdgcn_s_wait_asynccnt(N);
#endif
}

static __device__ __forceinline__ unsigned short f2bf(float f) {
  union { float f; unsigned int u; } v; v.f = f;
  unsigned int r = v.u + 0x7FFFu + ((v.u >> 16) & 1u);   // round-to-nearest-even
  return (unsigned short)(r >> 16);
}

static __device__ __forceinline__ bf16x16 frag16(const unsigned short* p0,
                                                 const unsigned short* p1) {
  bf16x8 lo = *reinterpret_cast<const bf16x8*>(p0);
  bf16x8 hi = *reinterpret_cast<const bf16x8*>(p1);
  return __builtin_shufflevector(lo, hi, 0, 1, 2, 3, 4, 5, 6, 7,
                                 8, 9, 10, 11, 12, 13, 14, 15);
}

// ---------------------------------------------------------------------------
// 1) Activation prep: Abig[b, 0:1024] = bf16(x), Abig[b, 1024:2048] = bf16(sx)
// ---------------------------------------------------------------------------
__global__ __launch_bounds__(256) void prep_act_kernel(
    const float* __restrict__ x, const float* __restrict__ sx,
    float* __restrict__ out_x, unsigned short* __restrict__ Abig) {
  size_t i = (size_t)blockIdx.x * 256 + threadIdx.x;   // over B*H
  size_t b = i >> 10;
  size_t h = i & 1023;
  float xv = x[i];
  float sv = sx[i];
  out_x[i] = xv;
  Abig[b * 2048 + h]        = f2bf(xv);
  Abig[b * 2048 + 1024 + h] = f2bf(sv);
}

// ---------------------------------------------------------------------------
// 2) Weight prep: fold time_mix coefficients into doubled-K bf16 weights.
// ---------------------------------------------------------------------------
__global__ __launch_bounds__(256) void prep_w_kernel(
    const float* __restrict__ Wk, const float* __restrict__ Wv,
    const float* __restrict__ Wr, const float* __restrict__ Wo,
    const float* __restrict__ tmk, const float* __restrict__ tmv,
    const float* __restrict__ tmr,
    unsigned short* __restrict__ Wkb, unsigned short* __restrict__ Wvb,
    unsigned short* __restrict__ Wrb, unsigned short* __restrict__ Wob) {
  size_t i = (size_t)blockIdx.x * 256 + threadIdx.x;   // over H*H
  size_t o = i >> 10;
  size_t h = i & 1023;
  float mk = tmk[h], mv = tmv[h], mr = tmr[h];
  float wk = Wk[i], wv = Wv[i], wr = Wr[i];
  size_t base = o * 2048 + h;
  Wkb[base]        = f2bf(wk * mk);
  Wkb[base + 1024] = f2bf(wk * (1.0f - mk));
  Wvb[base]        = f2bf(wv * mv);
  Wvb[base + 1024] = f2bf(wv * (1.0f - mv));
  Wrb[base]        = f2bf(wr * mr);
  Wrb[base + 1024] = f2bf(wr * (1.0f - mr));
  Wob[i] = f2bf(Wo[i]);
}

#define LDT 40   // padded LDS row stride in bf16 elements (16B-aligned chunks)

// ---------------------------------------------------------------------------
// 3) FUSED triple GEMM + WKV (k/v/r in-register, WKV epilogue).
//    Block tile M=128, N=64, BK=32. 8 waves as 4(M) x 2(N), 32x32 each,
//    3 accumulator sets = 12 WMMA per K-step per wave. Async LDS staging.
// ---------------------------------------------------------------------------
__global__ __launch_bounds__(256) void fused_gemm3_wkv_kernel(
    const unsigned short* __restrict__ A,   // [Bn x 2048] bf16
    const unsigned short* __restrict__ Wk,  // [Hn x 2048] bf16
    const unsigned short* __restrict__ Wv,
    const unsigned short* __restrict__ Wr,
    const float* __restrict__ stA, const float* __restrict__ stB,
    const float* __restrict__ stp,
    const float* __restrict__ tdec, const float* __restrict__ tfirst,
    float* __restrict__ oA, float* __restrict__ oB, float* __restrict__ oP,
    unsigned short* __restrict__ rwkv_bf) {
  const int K = 2048;
  __shared__ unsigned short sA[2][128 * LDT];   // 20 KB
  __shared__ unsigned short sK[2][64 * LDT];    // 10 KB
  __shared__ unsigned short sV[2][64 * LDT];    // 10 KB
  __shared__ unsigned short sR[2][64 * LDT];    // 10 KB

  const int tid  = threadIdx.x;
  const int lane = tid & 31;
  const int wave = tid >> 5;
  const int wm   = wave >> 1;      // 0..3  (M block of 32)
  const int wn   = wave & 1;       // 0..1  (N block of 32)
  const int l15  = lane & 15;
  const int lhi  = lane >> 4;      // 0/1

  const int bm = blockIdx.y * 128;
  const int bn = blockIdx.x * 64;

  const int sr0 = tid >> 2;          // 0..63
  const int sc  = (tid & 3) * 8;     // 0/8/16/24

  f32x8 acck[2][2], accv[2][2], accr[2][2];
#pragma unroll
  for (int i = 0; i < 2; ++i)
#pragma unroll
    for (int j = 0; j < 2; ++j) {
      f32x8 z = {0.f, 0.f, 0.f, 0.f, 0.f, 0.f, 0.f, 0.f};
      acck[i][j] = z; accv[i][j] = z; accr[i][j] = z;
    }

  // 5 x 16B per thread per K-step, direct to LDS via ASYNCcnt path.
  auto issue_stage = [&](int kt, int buf) {
    const unsigned short* Ap = A + (size_t)(bm + sr0) * K + (size_t)kt * 32 + sc;
    copy16_g2l(Ap,                   &sA[buf][(sr0)      * LDT + sc]);
    copy16_g2l(Ap + (size_t)64 * K,  &sA[buf][(sr0 + 64) * LDT + sc]);
    size_t woff = (size_t)(bn + sr0) * K + (size_t)kt * 32 + sc;
    copy16_g2l(Wk + woff, &sK[buf][sr0 * LDT + sc]);
    copy16_g2l(Wv + woff, &sV[buf][sr0 * LDT + sc]);
    copy16_g2l(Wr + woff, &sR[buf][sr0 * LDT + sc]);
  };

  const int nk = K >> 5;   // 64
  issue_stage(0, 0);
  for (int kt = 0; kt < nk; ++kt) {
    const int cur = kt & 1;
    if (kt + 1 < nk) {
      issue_stage(kt + 1, cur ^ 1);   // prefetch next stage (stays in flight)
      async_wait<5>();                // current stage's 5 ops have landed
    } else {
      async_wait<0>();
    }
    __syncthreads();                  // cross-wave visibility of LDS tiles

    const unsigned short* a_base = sA[cur];
    bf16x16 afrag[2];
#pragma unroll
    for (int i = 0; i < 2; ++i) {
      int row = wm * 32 + i * 16 + l15;
      afrag[i] = frag16(&a_base[row * LDT + lhi * 16],
                        &a_base[row * LDT + lhi * 16 + 8]);
    }
#pragma unroll
    for (int j = 0; j < 2; ++j) {
      int col = wn * 32 + j * 16 + l15;
      int off = col * LDT + lhi * 16;
      bf16x16 bk = frag16(&sK[cur][off], &sK[cur][off + 8]);
      bf16x16 bv = frag16(&sV[cur][off], &sV[cur][off + 8]);
      bf16x16 br = frag16(&sR[cur][off], &sR[cur][off + 8]);
#pragma unroll
      for (int i = 0; i < 2; ++i) {
        acck[i][j] = __builtin_amdgcn_wmma_f32_16x16x32_bf16(
            false, afrag[i], false, bk, (short)0, acck[i][j], false, false);
        accv[i][j] = __builtin_amdgcn_wmma_f32_16x16x32_bf16(
            false, afrag[i], false, bv, (short)0, accv[i][j], false, false);
        accr[i][j] = __builtin_amdgcn_wmma_f32_16x16x32_bf16(
            false, afrag[i], false, br, (short)0, accr[i][j], false, false);
      }
    }
    __syncthreads();                  // all waves done reading before rewrite
  }

  // ------------------- fused WKV epilogue -------------------
#pragma unroll
  for (int j = 0; j < 2; ++j) {
    int n = bn + wn * 32 + j * 16 + l15;
    float tf = tfirst[n];
    float td = tdec[n];
#pragma unroll
    for (int i = 0; i < 2; ++i) {
      int m0 = bm + wm * 32 + i * 16 + 8 * lhi;
#pragma unroll
      for (int r = 0; r < 8; ++r) {
        size_t idx = (size_t)(m0 + r) * Hn + n;
        float kk = acck[i][j][r];
        float vv = accv[i][j][r];
        float rp = accr[i][j][r];
        float sp = stp[idx], sAv = stA[idx], sBv = stB[idx];

        float rg = 1.0f / (1.0f + __expf(-rp));          // sigmoid(r_pre)
        float ww = tf + kk;
        float p  = fmaxf(sp, ww);
        float e1 = __expf(sp - p);
        float e2 = __expf(ww - p);
        float wkv  = (e1 * sAv + e2 * vv) / (e1 * sBv + e2 + 1e-8f);
        float rwkv = rg * wkv;

        float ww2 = sp + td;
        float p2  = fmaxf(ww2, kk);
        float e1n = __expf(ww2 - p2);
        float e2n = __expf(kk - p2);

        oA[idx] = e1n * sAv + e2n * vv;   // new_A
        oB[idx] = e1n * sBv + e2n;        // new_B
        oP[idx] = p2;                     // p2
        rwkv_bf[idx] = f2bf(rwkv);
      }
    }
  }
}

// ---------------------------------------------------------------------------
// 4) Plain tiled WMMA GEMM (out = rwkv @ Wo^T): block 128x128, 8 waves as
//    2x4, each 64x32, BK=32, async double-buffered LDS staging.
// ---------------------------------------------------------------------------
__global__ __launch_bounds__(256) void gemm_bf16_wmma_kernel(
    const unsigned short* __restrict__ A, const unsigned short* __restrict__ W,
    float* __restrict__ C, int K) {
  __shared__ unsigned short sA[2][128 * LDT];
  __shared__ unsigned short sW[2][128 * LDT];

  const int tid  = threadIdx.x;
  const int lane = tid & 31;
  const int wave = tid >> 5;
  const int wm   = wave >> 2;      // 0..1
  const int wn   = wave & 3;       // 0..3
  const int l15  = lane & 15;
  const int lhi  = lane >> 4;

  const int bm = blockIdx.y * 128;
  const int bn = blockIdx.x * 128;

  const int sr0 = tid >> 2;
  const int sc  = (tid & 3) * 8;

  f32x8 acc[4][2];
#pragma unroll
  for (int i = 0; i < 4; ++i)
#pragma unroll
    for (int j = 0; j < 2; ++j) {
      f32x8 z = {0.f, 0.f, 0.f, 0.f, 0.f, 0.f, 0.f, 0.f};
      acc[i][j] = z;
    }

  auto issue_stage = [&](int kt, int buf) {
    const unsigned short* Ap = A + (size_t)(bm + sr0) * K + (size_t)kt * 32 + sc;
    copy16_g2l(Ap,                  &sA[buf][(sr0)      * LDT + sc]);
    copy16_g2l(Ap + (size_t)64 * K, &sA[buf][(sr0 + 64) * LDT + sc]);
    const unsigned short* Wp = W + (size_t)(bn + sr0) * K + (size_t)kt * 32 + sc;
    copy16_g2l(Wp,                  &sW[buf][(sr0)      * LDT + sc]);
    copy16_g2l(Wp + (size_t)64 * K, &sW[buf][(sr0 + 64) * LDT + sc]);
  };

  const int nk = K >> 5;
  issue_stage(0, 0);
  for (int kt = 0; kt < nk; ++kt) {
    const int cur = kt & 1;
    if (kt + 1 < nk) {
      issue_stage(kt + 1, cur ^ 1);
      async_wait<4>();
    } else {
      async_wait<0>();
    }
    __syncthreads();

    const unsigned short* a_base = sA[cur];
    const unsigned short* w_base = sW[cur];

    bf16x16 afrag[4];
#pragma unroll
    for (int i = 0; i < 4; ++i) {
      int row = wm * 64 + i * 16 + l15;
      afrag[i] = frag16(&a_base[row * LDT + lhi * 16],
                        &a_base[row * LDT + lhi * 16 + 8]);
    }
    bf16x16 bfrag[2];
#pragma unroll
    for (int j = 0; j < 2; ++j) {
      int col = wn * 32 + j * 16 + l15;
      bfrag[j] = frag16(&w_base[col * LDT + lhi * 16],
                        &w_base[col * LDT + lhi * 16 + 8]);
    }

#pragma unroll
    for (int i = 0; i < 4; ++i)
#pragma unroll
      for (int j = 0; j < 2; ++j)
        acc[i][j] = __builtin_amdgcn_wmma_f32_16x16x32_bf16(
            false, afrag[i], false, bfrag[j], (short)0, acc[i][j], false, false);

    __syncthreads();
  }

#pragma unroll
  for (int i = 0; i < 4; ++i) {
    int m0 = bm + wm * 64 + i * 16 + lhi * 8;
#pragma unroll
    for (int j = 0; j < 2; ++j) {
      int n = bn + wn * 32 + j * 16 + l15;
#pragma unroll
      for (int r = 0; r < 8; ++r)
        C[(size_t)(m0 + r) * Hn + n] = acc[i][j][r];
    }
  }
}

// ---------------------------------------------------------------------------
// Host-side orchestration
// ---------------------------------------------------------------------------
extern "C" void kernel_launch(void* const* d_in, const int* in_sizes, int n_in,
                              void* d_out, int out_size, void* d_ws, size_t ws_size,
                              hipStream_t stream) {
  (void)in_sizes; (void)n_in; (void)out_size; (void)ws_size;

  const float* x     = (const float*)d_in[0];
  const float* st_x  = (const float*)d_in[1];
  const float* st_A  = (const float*)d_in[2];
  const float* st_B  = (const float*)d_in[3];
  const float* st_p  = (const float*)d_in[4];
  const float* tdec  = (const float*)d_in[5];
  const float* tfst  = (const float*)d_in[6];
  const float* tmk   = (const float*)d_in[7];
  const float* tmv   = (const float*)d_in[8];
  const float* tmr   = (const float*)d_in[9];
  const float* Wk    = (const float*)d_in[10];
  const float* Wv    = (const float*)d_in[11];
  const float* Wr    = (const float*)d_in[12];
  const float* Wo    = (const float*)d_in[13];

  float* out   = (float*)d_out;          // [out | x | new_A | new_B | p2]
  float* out_o = out;
  float* out_x = out + BHn;
  float* out_A = out + 2 * BHn;
  float* out_B = out + 3 * BHn;
  float* out_p = out + 4 * BHn;

  // Workspace layout (bf16 stored as u16), all offsets 256B aligned
  char* ws = (char*)d_ws;
  unsigned short* Abig = (unsigned short*)(ws);                       // 16384x2048
  unsigned short* Wkb  = (unsigned short*)(ws + 67108864);            // 1024x2048
  unsigned short* Wvb  = (unsigned short*)(ws + 71303168);
  unsigned short* Wrb  = (unsigned short*)(ws + 75497472);
  unsigned short* Wob  = (unsigned short*)(ws + 79691776);            // 1024x1024
  unsigned short* Rbf  = (unsigned short*)(ws + 81788928);            // 16384x1024

  const int nelem_blocks = (int)(BHn / 256);      // 65536
  const int w_blocks     = (Hn * Hn) / 256;       // 4096

  prep_act_kernel<<<nelem_blocks, 256, 0, stream>>>(x, st_x, out_x, Abig);
  prep_w_kernel<<<w_blocks, 256, 0, stream>>>(Wk, Wv, Wr, Wo, tmk, tmv, tmr,
                                              Wkb, Wvb, Wrb, Wob);

  dim3 fgrid(Hn / 64, Bn / 128);                  // (16, 128)
  fused_gemm3_wkv_kernel<<<fgrid, 256, 0, stream>>>(
      Abig, Wkb, Wvb, Wrb, st_A, st_B, st_p, tdec, tfst,
      out_A, out_B, out_p, Rbf);

  dim3 ggrid(Hn / 128, Bn / 128);                 // (8, 128)
  gemm_bf16_wmma_kernel<<<ggrid, 256, 0, stream>>>(Rbf, Wob, out_o, 1024);
}
